// SNNConv1d_1692217115112
// MI455X (gfx1250) — matrix operand order repaired and checked by reading, not compiled, not run on hardware
//
#include <hip/hip_runtime.h>

typedef __attribute__((ext_vector_type(16))) __bf16 v16bf;
typedef __attribute__((ext_vector_type(2)))  __bf16 v2bf;
typedef __attribute__((ext_vector_type(8)))  float  v8f;
typedef __attribute__((ext_vector_type(4)))  float  v4f;

constexpr int NB    = 16;    // batch
constexpr int CIN   = 32;
constexpr int LEN   = 1024;  // spatial length (OL == LEN with pad=1, k=3)
constexpr int NT    = 64;    // timesteps
constexpr int COUT  = 64;
constexpr int KW    = 3;
constexpr int OLBLK = 8;     // ol positions per block
constexpr int NLI   = OLBLK + 2;   // staged l-slices (halo of 1 each side)
constexpr int XPITCH = 36;   // padded cin pitch in halfs -> 18-bank stride, conflict free
constexpr float DECAY = 0.9f;
constexpr float THRESH = 0.25f;

__global__ __launch_bounds__(256)
void snn_conv1d_lif(const float* __restrict__ x,
                    const float* __restrict__ w,
                    float* __restrict__ out) {
  __shared__ __bf16 lds_x[NLI * NT * XPITCH];   // [li][t][cin]  ~45 KB
  __shared__ __bf16 lds_w[COUT * 96];           // [cout][kc], kc = k*32+cin  12 KB

  const int tid = threadIdx.x;
  const int blocksPerB = LEN / OLBLK;           // 128
  const int b   = blockIdx.x / blocksPerB;
  const int ol0 = (blockIdx.x % blocksPerB) * OLBLK;

  // ---- stage weights: global (cout, cin, k) -> lds_w[cout*96 + k*32 + cin] ----
  #pragma unroll
  for (int j = 0; j < (COUT * CIN * KW) / 256; ++j) {     // 24 iters
    int idx  = tid + j * 256;
    int cout = idx / (CIN * KW);
    int rem  = idx - cout * (CIN * KW);
    int cin  = rem / KW;
    int k    = rem - cin * KW;
    lds_w[cout * 96 + k * 32 + cin] = (__bf16)w[idx];
  }

  // ---- stage x slab: nontemporal float4 reads over t (512B/wave, coalesced),
  //      f32->bf16, LDS layout [li][t][cin] ----
  #pragma unroll
  for (int j = 0; j < (NLI * CIN * NT / 4) / 256; ++j) {  // 20 iters
    int idx = tid + j * 256;
    int tq  = idx & (NT / 4 - 1);                          // t quad: t = 4*tq
    int tmp = idx >> 4;
    int cin = tmp & (CIN - 1);
    int li  = tmp >> 5;                                    // 0..9
    int l   = ol0 - 1 + li;
    v4f v = {0.f, 0.f, 0.f, 0.f};
    if (l >= 0 && l < LEN) {
      const v4f* src = (const v4f*)x + ((((size_t)b * CIN + cin) * LEN + l) * (NT / 4) + tq);
      v = __builtin_nontemporal_load(src);                 // streamed: read-once data
    }
    __bf16* dst = &lds_x[(li * NT + 4 * tq) * XPITCH + cin];
    dst[0 * XPITCH] = (__bf16)v.x;
    dst[1 * XPITCH] = (__bf16)v.y;
    dst[2 * XPITCH] = (__bf16)v.z;
    dst[3 * XPITCH] = (__bf16)v.w;
  }
  __syncthreads();

  const int wave = tid >> 5;
  const int lane = tid & 31;
  const int lh   = lane >> 4;     // lane half
  const int m    = lane & 15;
  const int c0   = (wave & 3) * 16;   // this wave's cout chunk

  // ---- B fragments (weights), register-resident for the whole block ----
  // B, contiguous k order as A: lane n = cout, VGPR r slots {0,1} = K rows 2r,2r+1 (+16 for hi half)
  v16bf bfrag[3];
  {
    const __bf16* wrow = &lds_w[(c0 + m) * 96 + 16 * lh];
    #pragma unroll
    for (int s = 0; s < 3; ++s)
      #pragma unroll
      for (int r = 0; r < 8; ++r) {
        v2bf p = *(const v2bf*)(wrow + 32 * s + 2 * r);
        bfrag[s][2 * r]     = p.x;
        bfrag[s][2 * r + 1] = p.y;
      }
  }

  for (int pss = 0; pss < OLBLK / 2; ++pss) {
    const int olLocal = 2 * pss + (wave >> 2);    // 8 waves = 2 ol x 4 cout-chunks
    const int ol = ol0 + olLocal;

    v8f acc[4] = {};    // 4 t-tiles of act[t, cout16]

    #pragma unroll
    for (int i = 0; i < 4; ++i) {                 // t tile
      const int t = 16 * i + m;
      // Load ALL three k-step A-fragments first, then issue the 3 WMMAs:
      // lets the scheduler batch the DS loads instead of wait-dscnt(0) per WMMA.
      v16bf afr[3];
      #pragma unroll
      for (int s = 0; s < 3; ++s) {               // conv tap == WMMA k-step
        const int li = olLocal + s;               // l = ol + s - 1
        // A, contiguous k order as B: lane gives M=t row; VGPR v, slot -> K = 16*lh+2*v+slot
        const __bf16* xrow = &lds_x[(li * NT + t) * XPITCH + 16 * lh];
        #pragma unroll
        for (int v = 0; v < 8; ++v) {
          int cinb = 2 * v;
          v2bf q = *(const v2bf*)(xrow + cinb);
          afr[s][2 * v]     = q.x;
          afr[s][2 * v + 1] = q.y;
        }
      }
      #pragma unroll
      for (int s = 0; s < 3; ++s)
        acc[i] = __builtin_amdgcn_wmma_f32_16x16x32_bf16(
            false, afr[s], false, bfrag[s], (short)0, acc[i], false, false);
    }

    // ---- LIF scan over t, in registers. acc[i][r] = act at t = 16i + 8*lh + r ----
    const int cout = c0 + m;
    float* obase = out + ((((size_t)b) * COUT + cout) * LEN + ol) * NT;
    float pot = 0.f;                               // valid in low-half lanes
    #pragma unroll
    for (int i = 0; i < 4; ++i) {
      float p1 = pot;
      float sp1[8], sp2[8];
      #pragma unroll
      for (int r = 0; r < 8; ++r) {                // low half: t = 16i + r
        p1 = (p1 + acc[i][r]) * DECAY;
        float sk = (p1 >= THRESH) ? 1.f : 0.f;
        p1 = (sk > 0.f) ? 0.f : p1;
        sp1[r] = sk;
      }
      float ph = __shfl(p1, m);                    // carry low-half pot -> high half
      float p2 = ph;
      #pragma unroll
      for (int r = 0; r < 8; ++r) {                // high half: t = 16i + 8 + r
        p2 = (p2 + acc[i][r]) * DECAY;
        float sk = (p2 >= THRESH) ? 1.f : 0.f;
        p2 = (sk > 0.f) ? 0.f : p2;
        sp2[r] = sk;
      }
      pot = __shfl(p2, m + 16);                    // carry t=16i+15 -> next tile

      v4f v0, v1;
      if (lh == 0) {
        v0 = (v4f){sp1[0], sp1[1], sp1[2], sp1[3]};
        v1 = (v4f){sp1[4], sp1[5], sp1[6], sp1[7]};
      } else {
        v0 = (v4f){sp2[0], sp2[1], sp2[2], sp2[3]};
        v1 = (v4f){sp2[4], sp2[5], sp2[6], sp2[7]};
      }
      float* dst = obase + 16 * i + 8 * lh;        // 32B-aligned, t contiguous
      // Output is written once and never re-read: stream past L2 (TH=NT).
      __builtin_nontemporal_store(v0, (v4f*)dst);
      __builtin_nontemporal_store(v1, (v4f*)(dst + 4));
    }
  }
}

extern "C" void kernel_launch(void* const* d_in, const int* in_sizes, int n_in,
                              void* d_out, int out_size, void* d_ws, size_t ws_size,
                              hipStream_t stream) {
  const float* x = (const float*)d_in[0];   // (16, 32, 1024, 64) f32
  const float* w = (const float*)d_in[1];   // (64, 32, 3) f32
  float* out = (float*)d_out;               // (16, 64, 1024, 64) f32

  dim3 grid(NB * (LEN / OLBLK));            // 16 * 128 = 2048 blocks
  snn_conv1d_lif<<<grid, 256, 0, stream>>>(x, w, out);
}
